// SIGN_31808527794885
// MI455X (gfx1250) — matrix operand order repaired and checked by reading, not compiled, not run on hardware
//
#include <hip/hip_runtime.h>
#include <hip/hip_bf16.h>

// ---------------------------------------------------------------------------
// Types for CDNA5 WMMA (wave32, v_wmma_f32_16x16x32_bf16)
// ---------------------------------------------------------------------------
typedef __attribute__((ext_vector_type(16))) __bf16 v16bf_t;
typedef __attribute__((ext_vector_type(8)))  __bf16 v8bf_t;
typedef __attribute__((ext_vector_type(8)))  float  v8f_t;

#define GN     50000
#define GE     800000
#define GIN_C  128
#define GHID   256
#define GOUT_C 128
#define GCATC  768

// ---------------------------------------------------------------------------
// Utility kernels
// ---------------------------------------------------------------------------
__global__ void sign_zero_kernel(float* __restrict__ p, size_t n) {
    size_t i = (size_t)blockIdx.x * blockDim.x + threadIdx.x;
    size_t stride = (size_t)gridDim.x * blockDim.x;
    for (; i < n; i += stride) p[i] = 0.0f;
}

__global__ void sign_degree_kernel(float* __restrict__ deg,
                                   const int* __restrict__ col, int E) {
    int e = blockIdx.x * blockDim.x + threadIdx.x;
    if (e < E) atomicAdd(&deg[col[e]], 1.0f);
}

__global__ void sign_dis_kernel(float* __restrict__ dis,
                                const float* __restrict__ deg, int n) {
    int i = blockIdx.x * blockDim.x + threadIdx.x;
    if (i < n) {
        float d = deg[i];
        dis[i] = (d > 0.0f) ? rsqrtf(d) : 0.0f;
    }
}

// One warp (32 lanes) per edge, float4 per lane -> 128 channels.
// Gathers hit L2 (node table is 25.6MB << 192MB L2); scatter via f32 atomics.
__global__ void sign_gcn_kernel(float* __restrict__ out,
                                const float* __restrict__ x,
                                const int* __restrict__ row,
                                const int* __restrict__ col,
                                const float* __restrict__ dis, int E) {
    long long t = (long long)blockIdx.x * blockDim.x + threadIdx.x;
    int e = (int)(t >> 5);
    int q = (int)(t & 31);
    if (e >= E) return;
    int r = row[e];
    int c = col[e];
    float norm = dis[r] * dis[c];
    const float4 v = ((const float4*)(x + (size_t)r * GIN_C))[q];
    float* dst = out + (size_t)c * GIN_C + q * 4;
    atomicAdd(dst + 0, norm * v.x);
    atomicAdd(dst + 1, norm * v.y);
    atomicAdd(dst + 2, norm * v.z);
    atomicAdd(dst + 3, norm * v.w);
}

// ---------------------------------------------------------------------------
// BF16 WMMA GEMM:  C[M,Nw] (= or +=) A[M,K] @ W[K,Nw] + bias  (opt. ReLU)
//   Block: 256 threads (8 waves). C tile 128x64. Wave subtile 32x32
//   (2x2 v_wmma_f32_16x16x32_bf16 accumulators). K stepped by 32.
//   A/B staged to LDS as bf16 with padded pitch (40 elems = 80B, 16B aligned).
//   B staged transposed with K-pair packing -> ds_store_b32 (not b16).
// ---------------------------------------------------------------------------
#define BM 128
#define BN 64
#define BK 32
#define LDP 40   // BK + 8 pad (bf16 elements)

__device__ __forceinline__ unsigned pack_bf16(float lo, float hi) {
    union { __bf16 h[2]; unsigned u; } p;
    p.h[0] = (__bf16)lo;
    p.h[1] = (__bf16)hi;
    return p.u;
}

__global__ __launch_bounds__(256) void sign_wmma_gemm(
    const float* __restrict__ A, const float* __restrict__ W,
    const float* __restrict__ bias, float* __restrict__ C,
    int M, int K, int Nw, int ldC, int beta, int relu) {
    __shared__ __bf16 As[BM * LDP];
    __shared__ __bf16 Bs[BN * LDP];

    const int tid  = threadIdx.x;
    const int lane = tid & 31;
    const int wave = tid >> 5;
    const int wm   = wave & 3;   // wave row   (4 x 32 = 128)
    const int wn   = wave >> 2;  // wave col   (2 x 32 = 64)
    const int blockM = blockIdx.x * BM;
    const int blockN = blockIdx.y * BN;

    v8f_t acc[2][2] = {};

    // A staging: 2 threads per row, 16 floats each (contiguous, 64B aligned).
    const int arow = tid >> 1;
    const int acol = (tid & 1) * 16;
    // B staging: thread t handles K-pair k=2*(t>>4), 4 consecutive n from
    // nbase=(t&15)*4. 16 consecutive lanes read 256B contiguous of one W row.
    const int bk = (tid >> 4) * 2;   // 0,2,...,30
    const int bn = (tid & 15) * 4;   // 0,4,...,60

    for (int k0 = 0; k0 < K; k0 += BK) {
        // ---- load A tile (fp32 -> bf16), ds_store_b128 x2 ----
        {
            const int gr = blockM + arow;
            __bf16* dst = &As[arow * LDP + acol];
            v8bf_t o0, o1;
            if (gr < M) {
                const float4* s4 = (const float4*)(A + (size_t)gr * K + k0 + acol);
                float4 a0 = s4[0], a1 = s4[1], a2 = s4[2], a3 = s4[3];
                o0[0] = (__bf16)a0.x; o0[1] = (__bf16)a0.y; o0[2] = (__bf16)a0.z; o0[3] = (__bf16)a0.w;
                o0[4] = (__bf16)a1.x; o0[5] = (__bf16)a1.y; o0[6] = (__bf16)a1.z; o0[7] = (__bf16)a1.w;
                o1[0] = (__bf16)a2.x; o1[1] = (__bf16)a2.y; o1[2] = (__bf16)a2.z; o1[3] = (__bf16)a2.w;
                o1[4] = (__bf16)a3.x; o1[5] = (__bf16)a3.y; o1[6] = (__bf16)a3.z; o1[7] = (__bf16)a3.w;
            } else {
                #pragma unroll
                for (int i = 0; i < 8; ++i) { o0[i] = (__bf16)0.0f; o1[i] = (__bf16)0.0f; }
            }
            *(v8bf_t*)(dst)     = o0;
            *(v8bf_t*)(dst + 8) = o1;
        }
        // ---- load B tile transposed: Bs[n][k] = W[k0+k][blockN+n] ----
        // Pack (k,k+1) bf16 pair -> one 32-bit DS store per (n).
        {
            const float4* r0 = (const float4*)(W + (size_t)(k0 + bk)     * Nw + blockN + bn);
            const float4* r1 = (const float4*)(W + (size_t)(k0 + bk + 1) * Nw + blockN + bn);
            float4 w0 = r0[0];
            float4 w1 = r1[0];
            unsigned* base = (unsigned*)&Bs[bn * LDP + bk];
            const int strideU = LDP / 2;   // 20 dwords per n-row (bk even -> aligned)
            base[0 * strideU] = pack_bf16(w0.x, w1.x);
            base[1 * strideU] = pack_bf16(w0.y, w1.y);
            base[2 * strideU] = pack_bf16(w0.z, w1.z);
            base[3 * strideU] = pack_bf16(w0.w, w1.w);
        }
        __syncthreads();

        // ---- build fragments per ISA 7.12.2 layouts ----
        const int mrow  = lane & 15;
        const int khalf = lane >> 4;
        v16bf_t afrag[2], bfrag[2];
        #pragma unroll
        for (int im = 0; im < 2; ++im) {
            // A 16x32 bf16: lane=M; elems 0..7 -> K=khalf*8+0..7,
            //               elems 8..15 -> K=16+khalf*8+0..7
            const __bf16* ap = &As[(wm * 32 + im * 16 + mrow) * LDP];
            v8bf_t lo = *(const v8bf_t*)(ap + khalf * 8);
            v8bf_t hi = *(const v8bf_t*)(ap + 16 + khalf * 8);
            #pragma unroll
            for (int i = 0; i < 8; ++i) { afrag[im][i] = lo[i]; afrag[im][i + 8] = hi[i]; }
        }
        #pragma unroll
        for (int in = 0; in < 2; ++in) {
            // B 32x16 bf16: lane=N column; elems 0..7 -> K=khalf*8+0..7,
            //               elems 8..15 -> K=16+khalf*8+0..7
            const __bf16* bp = &Bs[(wn * 32 + in * 16 + mrow) * LDP];
            v8bf_t lo = *(const v8bf_t*)(bp + khalf * 8);
            v8bf_t hi = *(const v8bf_t*)(bp + 16 + khalf * 8);
            #pragma unroll
            for (int i = 0; i < 8; ++i) { bfrag[in][i] = lo[i]; bfrag[in][i + 8] = hi[i]; }
        }

        #pragma unroll
        for (int im = 0; im < 2; ++im)
            #pragma unroll
            for (int in = 0; in < 2; ++in)
                acc[im][in] = __builtin_amdgcn_wmma_f32_16x16x32_bf16(
                    false, afrag[im], false, bfrag[in],
                    (short)0, acc[im][in], false, false);

        __syncthreads();
    }

    // ---- epilogue: bias (+C) (ReLU) ----
    // C/D f32 16x16: lane -> N=(lane&15); VGPR r -> M=(lane>>4)*8 + r
    const int nloc = lane & 15;
    const int mgrp = (lane >> 4) * 8;
    const bool fullTile = (blockM + BM) <= M;
    #pragma unroll
    for (int im = 0; im < 2; ++im) {
        #pragma unroll
        for (int in = 0; in < 2; ++in) {
            const int n = blockN + wn * 32 + in * 16 + nloc;
            const float bv = bias[n];
            if (fullTile) {
                #pragma unroll
                for (int r = 0; r < 8; ++r) {
                    const int m = blockM + wm * 32 + im * 16 + mgrp + r;
                    float v = acc[im][in][r] + bv;
                    const size_t idx = (size_t)m * ldC + n;
                    if (beta) v += C[idx];
                    if (relu) v = fmaxf(v, 0.0f);
                    C[idx] = v;
                }
            } else {
                #pragma unroll
                for (int r = 0; r < 8; ++r) {
                    const int m = blockM + wm * 32 + im * 16 + mgrp + r;
                    if (m < M) {
                        float v = acc[im][in][r] + bv;
                        const size_t idx = (size_t)m * ldC + n;
                        if (beta) v += C[idx];
                        if (relu) v = fmaxf(v, 0.0f);
                        C[idx] = v;
                    }
                }
            }
        }
    }
}

// ---------------------------------------------------------------------------
// Host-side orchestration
// ---------------------------------------------------------------------------
static inline void launch_gemm(const float* A, const float* W, const float* b,
                               float* C, int M, int K, int Nw, int ldC,
                               int beta, int relu, hipStream_t s) {
    dim3 grid((M + BM - 1) / BM, Nw / BN);
    sign_wmma_gemm<<<grid, 256, 0, s>>>(A, W, b, C, M, K, Nw, ldC, beta, relu);
}

extern "C" void kernel_launch(void* const* d_in, const int* in_sizes, int n_in,
                              void* d_out, int out_size, void* d_ws, size_t ws_size,
                              hipStream_t stream) {
    (void)in_sizes; (void)n_in; (void)out_size; (void)ws_size;

    const float* x   = (const float*)d_in[0];
    const int*   ei  = (const int*)d_in[1];
    const int*   row = ei;        // edge_index[0]
    const int*   col = ei + GE;   // edge_index[1]

    auto F = [&](int i) { return (const float*)d_in[i]; };
    // Param order (setup_inputs insertion order):
    //   2 + 6*b : branch b -> W0, b0, W1, b1, Wr, br
    //   20      : merger   -> W0, b0, W1, b1, Wr, br

    // Workspace layout (floats): ~257 MB total
    float* ws  = (float*)d_ws;
    size_t off = 0;
    float* deg = ws + off; off += GN;
    float* dis = ws + off; off += GN;
    float* x1  = ws + off; off += (size_t)GN * GIN_C;
    float* x2  = ws + off; off += (size_t)GN * GIN_C;
    float* h   = ws + off; off += (size_t)GN * GHID;
    float* cat = ws + off; off += (size_t)GN * GCATC;

    // ---- degree normalization (computed fresh every call: deterministic) ----
    sign_zero_kernel<<<256, 256, 0, stream>>>(deg, (size_t)GN);
    sign_degree_kernel<<<(GE + 255) / 256, 256, 0, stream>>>(deg, col, GE);
    sign_dis_kernel<<<(GN + 255) / 256, 256, 0, stream>>>(dis, deg, GN);

    const float* xin[3] = { x, x1, x2 };
    for (int b = 0; b < 3; ++b) {
        if (b > 0) {
            float* dst       = (b == 1) ? x1 : x2;
            const float* src = (b == 1) ? x  : x1;
            sign_zero_kernel<<<4096, 256, 0, stream>>>(dst, (size_t)GN * GIN_C);
            const long long tot = (long long)GE * 32;
            sign_gcn_kernel<<<(unsigned)((tot + 255) / 256), 256, 0, stream>>>(
                dst, src, row, col, dis, GE);
        }
        const int base = 2 + 6 * b;
        const float* Xb = xin[b];
        float* catb = cat + (size_t)b * GHID;   // column slice, ldC = 768
        // res = Xb @ Wr + br            -> cat slice
        launch_gemm(Xb, F(base + 4), F(base + 5), catb, GN, GIN_C, GHID, GCATC, 0, 0, stream);
        // h = relu(Xb @ W0 + b0)
        launch_gemm(Xb, F(base + 0), F(base + 1), h,    GN, GIN_C, GHID, GHID,  0, 1, stream);
        // cat slice += h @ W1 + b1
        launch_gemm(h,  F(base + 2), F(base + 3), catb, GN, GHID,  GHID, GCATC, 1, 0, stream);
    }

    // ---- merger MLP on cat [N, 768] ----
    const int mb = 20;
    float* out = (float*)d_out;
    // out = cat @ Wr + br
    launch_gemm(cat, F(mb + 4), F(mb + 5), out, GN, GCATC, GOUT_C, GOUT_C, 0, 0, stream);
    // h = relu(cat @ W0 + b0)
    launch_gemm(cat, F(mb + 0), F(mb + 1), h,   GN, GCATC, GHID,   GHID,   0, 1, stream);
    // out += h @ W1 + b1
    launch_gemm(h,   F(mb + 2), F(mb + 3), out, GN, GHID,  GOUT_C, GOUT_C, 1, 0, stream);
}
